// DITBackbone_12824772346095
// MI455X (gfx1250) — matrix operand order repaired and checked by reading, not compiled, not run on hardware
//
#include <hip/hip_runtime.h>
#include <hip/hip_bf16.h>

// ---------------- constants ----------------
#define CB   2          // batch
#define CN   1024       // half sequence
#define CS   2048       // full sequence (2N)
#define CT   4096       // total rows (B * 2N)
#define CDIM 768
#define CH   12
#define CHD  64
#define CL   12
#define CVOC 32000
#define CMLP 3072

typedef __attribute__((ext_vector_type(16))) __bf16 v16bf;
typedef __attribute__((ext_vector_type(8)))  float  v8f;
typedef __attribute__((ext_vector_type(4)))  float  v4f;
typedef __attribute__((ext_vector_type(4)))  unsigned int v4u;
typedef __attribute__((ext_vector_type(8)))  int    v8i;
typedef __attribute__((ext_vector_type(4)))  int    v4i;

__device__ __forceinline__ unsigned short f2bf(float x) {
    unsigned int u = __float_as_uint(x);
    unsigned int r = u + 0x7FFFu + ((u >> 16) & 1u);
    return (unsigned short)(r >> 16);
}
__device__ __forceinline__ float gelu_tanh(float x) {
    float x3 = x * x * x;
    return 0.5f * x * (1.f + tanhf(0.7978845608028654f * (x + 0.044715f * x3)));
}
__device__ __forceinline__ float silu(float x) { return x / (1.f + __expf(-x)); }

// ---------------- timestep embedding -> c = silu(temb) ----------------
__global__ __launch_bounds__(256) void temb_kernel(
    const float* __restrict__ sigma, const float* __restrict__ w1,
    const float* __restrict__ b1, const float* __restrict__ w2,
    const float* __restrict__ b2, float* __restrict__ cbuf) {
    int b = blockIdx.x, tid = threadIdx.x;
    __shared__ float se[256];
    __shared__ float st[768];
    float fr = __expf(-(9.2103403719761836f / 128.f) * (float)(tid & 127));
    float arg = sigma[b] * fr;
    se[tid] = (tid < 128) ? cosf(arg) : sinf(arg);
    __syncthreads();
    for (int j = tid; j < 768; j += 256) {
        float acc = b1[j];
        for (int k = 0; k < 256; ++k) acc += se[k] * w1[k * 768 + j];
        st[j] = silu(acc);
    }
    __syncthreads();
    for (int j = tid; j < 768; j += 256) {
        float acc = b2[j];
        for (int k = 0; k < 768; ++k) acc += st[k] * w2[k * 768 + j];
        cbuf[b * 768 + j] = silu(acc);
    }
}

// ---------------- adaLN projections ----------------
__global__ __launch_bounds__(256) void ada_kernel(
    const float* __restrict__ c, const float* __restrict__ aw,
    const float* __restrict__ ab, float* __restrict__ out) {
    int g = blockIdx.x * 256 + threadIdx.x;        // over L*B*4608
    int l = g / (CB * 4608);
    int rem = g % (CB * 4608);
    int b = rem / 4608, j = rem % 4608;
    float acc = ab[l * 4608 + j];
    const float* cb = c + b * 768;
    const float* wl = aw + (size_t)l * 768 * 4608 + j;
    for (int k = 0; k < 768; ++k) acc += cb[k] * wl[(size_t)k * 4608];
    out[g] = acc;
}

__global__ __launch_bounds__(256) void fin_ada_kernel(
    const float* __restrict__ c, const float* __restrict__ aw,
    const float* __restrict__ ab, float* __restrict__ out) {
    int g = blockIdx.x * 256 + threadIdx.x;        // over B*1536
    int b = g / 1536, j = g % 1536;
    float acc = ab[j];
    const float* cb = c + b * 768;
    for (int k = 0; k < 768; ++k) acc += cb[k] * aw[(size_t)k * 1536 + j];
    out[g] = acc;
}

// ---------------- embedding gather ----------------
__global__ __launch_bounds__(256) void gather_kernel(
    const int* __restrict__ idx, const float* __restrict__ embed,
    float* __restrict__ x) {
    int g = blockIdx.x * 256 + threadIdx.x;        // over T*DIM
    int r = g / CDIM, d = g - r * CDIM;
    x[g] = embed[(size_t)idx[r] * CDIM + d];
}

// ---------------- fused LN + adaLN modulate, emit bf16 ----------------
__global__ __launch_bounds__(256) void ln_mod_kernel(
    const float* __restrict__ x, const float* __restrict__ w,
    const float* __restrict__ ada, int shiftOff, int scaleOff, int adaStride,
    unsigned short* __restrict__ out) {
    int row = blockIdx.x, tid = threadIdx.x;
    int b = row >> 11;                             // 2048 rows per batch
    __shared__ float red[256];
    const float* xr = x + (size_t)row * CDIM;
    float v0 = xr[tid], v1 = xr[tid + 256], v2 = xr[tid + 512];
    red[tid] = v0 + v1 + v2;
    __syncthreads();
    for (int off = 128; off > 0; off >>= 1) {
        if (tid < off) red[tid] += red[tid + off];
        __syncthreads();
    }
    float mu = red[0] * (1.f / 768.f);
    __syncthreads();
    float d0 = v0 - mu, d1 = v1 - mu, d2 = v2 - mu;
    red[tid] = d0 * d0 + d1 * d1 + d2 * d2;
    __syncthreads();
    for (int off = 128; off > 0; off >>= 1) {
        if (tid < off) red[tid] += red[tid + off];
        __syncthreads();
    }
    float rstd = rsqrtf(red[0] * (1.f / 768.f) + 1e-5f);
    const float* ap = ada + (size_t)b * adaStride;
    unsigned short* orow = out + (size_t)row * CDIM;
    int c0 = tid, c1 = tid + 256, c2 = tid + 512;
    orow[c0] = f2bf(d0 * rstd * w[c0] * (1.f + ap[scaleOff + c0]) + ap[shiftOff + c0]);
    orow[c1] = f2bf(d1 * rstd * w[c1] * (1.f + ap[scaleOff + c1]) + ap[shiftOff + c1]);
    orow[c2] = f2bf(d2 * rstd * w[c2] * (1.f + ap[scaleOff + c2]) + ap[shiftOff + c2]);
}

// ---------------- RoPE in-place on q,k of qkv ----------------
__global__ __launch_bounds__(256) void rope_kernel(float* __restrict__ qkv) {
    int g = blockIdx.x * 256 + threadIdx.x;        // over T*768 pair-elements
    int r = g / 768;
    int rem = g - r * 768;
    int e = rem / 384;                             // 0=q, 1=k
    int hd = rem - e * 384;
    int hh = hd >> 5, d = hd & 31;
    int col = e * CDIM + hh * CHD + d;
    size_t off = (size_t)r * (3 * CDIM);
    int t = r & (CS - 1);
    int p = t & (CN - 1);
    float inv = __expf(-(9.2103403719761836f / 32.f) * (float)d);
    float ang = (float)p * inv;
    float cv = cosf(ang), sv = sinf(ang);
    float a = qkv[off + col];
    float bv = qkv[off + col + 32];
    qkv[off + col]      = a * cv - bv * sv;
    qkv[off + col + 32] = bv * cv + a * sv;
}

// ---------------- flash attention with block-diffusion mask ----------------
__global__ __launch_bounds__(256) void attn_kernel(
    const float* __restrict__ qkv, unsigned short* __restrict__ obf) {
    const int qb = blockIdx.x;  // 0..127  (16-row query block)
    const int h  = blockIdx.y;
    const int b  = blockIdx.z;
    const int tid = threadIdx.x;
    __shared__ float qs[16 * 65], ks[16 * 65], vs[16 * 65], accs[16 * 65];
    __shared__ float sc[16 * 16];
    __shared__ float mrow[16], lrow[16], cfrow[16];

    for (int t = 0; t < 4; ++t) {
        int idx = tid + t * 256;
        int qi = idx >> 6, d = idx & 63;
        int grow = b * CS + qb * 16 + qi;
        qs[qi * 65 + d] = qkv[(size_t)grow * (3 * CDIM) + h * CHD + d] * 0.125f;
        accs[qi * 65 + d] = 0.f;
    }
    if (tid < 16) { mrow[tid] = -3.0e38f; lrow[tid] = 0.f; }
    __syncthreads();

    int nkb = (qb < 64) ? (qb + 1) : (qb - 63);
    for (int it = 0; it < nkb; ++it) {
        int kb = (qb < 64) ? ((it == 0) ? qb : (63 + it)) : (64 + it);
        for (int t = 0; t < 4; ++t) {
            int idx = tid + t * 256;
            int kj = idx >> 6, d = idx & 63;
            size_t grow = (size_t)(b * CS + kb * 16 + kj) * (3 * CDIM) + h * CHD + d;
            ks[kj * 65 + d] = qkv[grow + CDIM];
            vs[kj * 65 + d] = qkv[grow + 2 * CDIM];
        }
        __syncthreads();
        {
            int qi = tid & 15, kj = tid >> 4;
            float s = 0.f;
            for (int d = 0; d < 64; ++d) s += qs[qi * 65 + d] * ks[kj * 65 + d];
            sc[qi * 16 + kj] = s;
        }
        __syncthreads();
        if (tid < 16) {
            int qi = tid;
            float mold = mrow[qi], mx = mold;
            for (int j = 0; j < 16; ++j) mx = fmaxf(mx, sc[qi * 16 + j]);
            float csum = 0.f;
            for (int j = 0; j < 16; ++j) {
                float p = __expf(sc[qi * 16 + j] - mx);
                sc[qi * 16 + j] = p;
                csum += p;
            }
            float cf = __expf(mold - mx);
            lrow[qi] = lrow[qi] * cf + csum;
            cfrow[qi] = cf;
            mrow[qi] = mx;
        }
        __syncthreads();
        {
            int qi = tid & 15, g = tid >> 4;
            for (int dd = 0; dd < 4; ++dd) {
                int d = g * 4 + dd;
                float a = accs[qi * 65 + d] * cfrow[qi];
                for (int j = 0; j < 16; ++j) a += sc[qi * 16 + j] * vs[j * 65 + d];
                accs[qi * 65 + d] = a;
            }
        }
        __syncthreads();
    }
    for (int t = 0; t < 4; ++t) {
        int idx = tid + t * 256;
        int qi = idx >> 6, d = idx & 63;
        int grow = b * CS + qb * 16 + qi;
        obf[(size_t)grow * CDIM + h * CHD + d] = f2bf(accs[qi * 65 + d] / lrow[qi]);
    }
}

// ---------------- WMMA bf16 GEMM ----------------
// Block tile 128x128, K-step 32, double-buffered LDS, 8 waves,
// each wave computes 32(M) x 64(N) = 2x4 wmma tiles per K-step.
// A tile (bf16, uniform base) is staged by the Tensor Data Mover with LDS
// padding 16B per 64B row -> pitch 80B (= ASTR 40 ushorts). B tile (f32)
// is converted to bf16 in registers and stored transposed via ds_store_b128.
// EPI 0: Cf = acc (+bias)                   (f32 out)
// EPI 1: Cb = f2bf(gelu(acc+bias))          (bf16 out)
// EPI 2: Cf = Cf + gate[b,n]*(acc (+bias))  (residual-gate, f32 in-place)
union FragU { v4u q[2]; v16bf v; };

template <int EPI, bool HB>
__global__ __launch_bounds__(256) void gemm_bf16_kernel(
    const unsigned short* __restrict__ A, const float* __restrict__ Bw,
    float* __restrict__ Cf, unsigned short* __restrict__ Cb,
    const float* __restrict__ bias, const float* __restrict__ gate,
    int M, int N, int K, int gateStride) {
    constexpr int ASTR = 40, BSTR = 40;
    __shared__ __align__(16) unsigned short As[2][128 * ASTR];
    __shared__ __align__(16) unsigned short Bs[2][128 * BSTR];

    const int tid = threadIdx.x;
    const int lane = tid & 31;
    const int wid = tid >> 5;
    const int wm = wid >> 1, wn = wid & 1;
    const int hf = lane >> 4, lanelo = lane & 15;
    const int rowBase = blockIdx.y * 128;
    const int colBase = blockIdx.x * 128;

    // ---- TDM descriptor pieces (uniform) ----
    const unsigned ldsA0 = (unsigned)(size_t)&As[0][0];
    const unsigned ldsA1 = (unsigned)(size_t)&As[1][0];
    const size_t aBase = (size_t)(A + (size_t)rowBase * K);   // byte address
    // group1: data_size=2B, pad_enable, pad_interval=16 dwords (code 3),
    // pad_amount=4 dwords (code 3), tile 32 x 128, tensor dims K x M,
    // dim0 stride = K elements.
    const unsigned g1d0 = (1u << 16) | (1u << 20) | (3u << 22) | (3u << 25);
    const unsigned g1d1 = ((unsigned)K & 0xffffu) << 16;
    const unsigned g1d2 = (((unsigned)K >> 16) & 0xffffu) |
                          (((unsigned)M & 0xffffu) << 16);
    const unsigned g1d3 = (((unsigned)M >> 16) & 0xffffu) | (32u << 16);
    const unsigned g1d4 = 128u;
    const unsigned g1d5 = (unsigned)K;

    auto stageA = [&](int bsel, size_t gaddr) {
        if (wid == 0) {
            unsigned lds = bsel ? ldsA1 : ldsA0;
            v4u g0 = {1u, lds, (unsigned)gaddr,
                      (unsigned)((gaddr >> 32) & 0x01FFFFFFu) | 0x80000000u};
            v8i g1 = {(int)g1d0, (int)g1d1, (int)g1d2, (int)g1d3,
                      (int)g1d4, (int)g1d5, 0, 0};
            v4i g2 = {1, 1, 0, 0};   // tensor_dim2=1, tensor_dim3=1, tile_dim3=0
            v4i g3 = {0, 0, 0, 0};
            v8i g4 = {0, 0, 0, 0, 0, 0, 0, 0};
            __builtin_amdgcn_tensor_load_to_lds(g0, g1, g2, g3, g4, 0);
        }
    };

    // B staging assignment: 128 cols x 2 16-k halves, incremental pointer
    const int bN = tid & 127, bKh = tid >> 7;
    const float* bptr = Bw + (size_t)(bKh * 16) * N + colBase + bN;
    auto stageB = [&](int bsel) {
        float f[16];
#pragma unroll
        for (int kk = 0; kk < 16; ++kk) f[kk] = bptr[(size_t)kk * N];
        unsigned int w[8];
#pragma unroll
        for (int i = 0; i < 8; ++i)
            w[i] = (unsigned int)f2bf(f[2 * i]) |
                   ((unsigned int)f2bf(f[2 * i + 1]) << 16);
        v4u q0 = {w[0], w[1], w[2], w[3]};
        v4u q1 = {w[4], w[5], w[6], w[7]};
        unsigned short* bdst = &Bs[bsel][bN * BSTR + bKh * 16];
        *(v4u*)bdst       = q0;
        *(v4u*)(bdst + 8) = q1;
        bptr += (size_t)32 * N;
    };

    v8f acc[2][4];
#pragma unroll
    for (int i = 0; i < 2; ++i)
#pragma unroll
        for (int j = 0; j < 4; ++j)
#pragma unroll
            for (int v = 0; v < 8; ++v) acc[i][j][v] = 0.f;

    const int nK = K >> 5;
    stageB(0);
    stageA(0, aBase);
    for (int kt = 0; kt < nK; ++kt) {
        if (wid == 0) __builtin_amdgcn_s_wait_tensorcnt(0);
        __syncthreads();
        int cur = kt & 1;
        if (kt + 1 < nK) {
            stageB(cur ^ 1);
            stageA(cur ^ 1, aBase + (size_t)(kt + 1) * 64);
        }

        FragU fa[2], fb[4];
#pragma unroll
        for (int mi = 0; mi < 2; ++mi) {
            int arow = wm * 32 + mi * 16 + lanelo;
            fa[mi].q[0] = *(const v4u*)&As[cur][arow * ASTR + hf * 8];
            fa[mi].q[1] = *(const v4u*)&As[cur][arow * ASTR + 16 + hf * 8];
        }
#pragma unroll
        for (int ni = 0; ni < 4; ++ni) {
            int brow = wn * 64 + ni * 16 + lanelo;
            fb[ni].q[0] = *(const v4u*)&Bs[cur][brow * BSTR + hf * 8];
            fb[ni].q[1] = *(const v4u*)&Bs[cur][brow * BSTR + 16 + hf * 8];
        }
#pragma unroll
        for (int mi = 0; mi < 2; ++mi)
#pragma unroll
            for (int ni = 0; ni < 4; ++ni)
                acc[mi][ni] = __builtin_amdgcn_wmma_f32_16x16x32_bf16(
                    false, fa[mi].v, false, fb[ni].v, (short)0, acc[mi][ni],
                    false, false);
    }

    // epilogue: per (mi,ni) the 8 rows are an 8-aligned run -> batch index,
    // bias and gate are loop-invariant; stride pointers by N per row.
#pragma unroll
    for (int mi = 0; mi < 2; ++mi)
#pragma unroll
        for (int ni = 0; ni < 4; ++ni) {
            int r0 = rowBase + wm * 32 + mi * 16 + hf * 8;
            int cidx = colBase + wn * 64 + ni * 16 + lanelo;
            float bv = HB ? bias[cidx] : 0.f;
            if (EPI == 1) {
                unsigned short* cp = Cb + (size_t)r0 * N + cidx;
#pragma unroll
                for (int v = 0; v < 8; ++v) {
                    *cp = f2bf(gelu_tanh(acc[mi][ni][v] + bv));
                    cp += N;
                }
            } else if (EPI == 2) {
                int bb = r0 >> 11;
                float gv = gate[(size_t)bb * gateStride + cidx];
                float* cp = Cf + (size_t)r0 * N + cidx;
#pragma unroll
                for (int v = 0; v < 8; ++v) {
                    *cp = *cp + gv * (acc[mi][ni][v] + bv);
                    cp += N;
                }
            } else {
                float* cp = Cf + (size_t)r0 * N + cidx;
#pragma unroll
                for (int v = 0; v < 8; ++v) {
                    *cp = acc[mi][ni][v] + bv;
                    cp += N;
                }
            }
        }
}

// ---------------- host launcher ----------------
extern "C" void kernel_launch(void* const* d_in, const int* in_sizes, int n_in,
                              void* d_out, int out_size, void* d_ws,
                              size_t ws_size, hipStream_t stream) {
    (void)in_sizes; (void)n_in; (void)out_size; (void)ws_size;

    const int*   indices   = (const int*)d_in[0];
    const float* sigma     = (const float*)d_in[1];
    const float* embed     = (const float*)d_in[2];
    const float* t_w1      = (const float*)d_in[3];
    const float* t_b1      = (const float*)d_in[4];
    const float* t_w2      = (const float*)d_in[5];
    const float* t_b2      = (const float*)d_in[6];
    const float* norm1_w   = (const float*)d_in[7];
    const float* Wqkv      = (const float*)d_in[8];
    const float* Wout      = (const float*)d_in[9];
    const float* norm2_w   = (const float*)d_in[10];
    const float* mlp_w1    = (const float*)d_in[11];
    const float* mlp_b1    = (const float*)d_in[12];
    const float* mlp_w2    = (const float*)d_in[13];
    const float* mlp_b2    = (const float*)d_in[14];
    const float* ada_w     = (const float*)d_in[15];
    const float* ada_b     = (const float*)d_in[16];
    const float* fin_norm  = (const float*)d_in[17];
    const float* fin_w     = (const float*)d_in[18];
    const float* fin_b     = (const float*)d_in[19];
    const float* fin_ada_w = (const float*)d_in[20];
    const float* fin_ada_b = (const float*)d_in[21];
    float* out = (float*)d_out;

    // workspace layout (bytes, 256-aligned cumulative)
    char* wsb = (char*)d_ws;
    float*          cbuf    = (float*)(wsb + 0);                 //   6144
    float*          ada_all = (float*)(wsb + 6144);              // 442368
    float*          finada  = (float*)(wsb + 448512);            //  12288
    float*          xbuf    = (float*)(wsb + 460800);            // 12.58M
    unsigned short* hbuf    = (unsigned short*)(wsb + 13043712); //  6.29M
    unsigned short* obuf    = (unsigned short*)(wsb + 19335168); //  6.29M
    unsigned short* m1buf   = (unsigned short*)(wsb + 25626624); // 25.17M
    float*          qkvbuf  = (float*)(wsb + 50792448);          // 37.75M

    temb_kernel<<<CB, 256, 0, stream>>>(sigma, t_w1, t_b1, t_w2, t_b2, cbuf);
    ada_kernel<<<(CL * CB * 4608) / 256, 256, 0, stream>>>(cbuf, ada_w, ada_b, ada_all);
    fin_ada_kernel<<<(CB * 1536) / 256, 256, 0, stream>>>(cbuf, fin_ada_w, fin_ada_b, finada);
    gather_kernel<<<(CT * CDIM) / 256, 256, 0, stream>>>(indices, embed, xbuf);

    for (int l = 0; l < CL; ++l) {
        const float* ada_l = ada_all + (size_t)l * CB * 4608;
        // h = ln(x)*(1+scm) + sm   -> hbuf (bf16)
        ln_mod_kernel<<<CT, 256, 0, stream>>>(xbuf, norm1_w + l * CDIM, ada_l,
                                              0, 768, 4608, hbuf);
        // qkv = h @ Wqkv[l]
        gemm_bf16_kernel<0, false><<<dim3((3 * CDIM) / 128, CT / 128), 256, 0, stream>>>(
            hbuf, Wqkv + (size_t)l * CDIM * 3 * CDIM, qkvbuf, nullptr, nullptr,
            nullptr, CT, 3 * CDIM, CDIM, 0);
        // rope on q,k
        rope_kernel<<<(CT * 768) / 256, 256, 0, stream>>>(qkvbuf);
        // attention -> obuf (bf16)
        attn_kernel<<<dim3(CS / 16, CH, CB), 256, 0, stream>>>(qkvbuf, obuf);
        // x += gm * (o @ Wout[l])
        gemm_bf16_kernel<2, false><<<dim3(CDIM / 128, CT / 128), 256, 0, stream>>>(
            obuf, Wout + (size_t)l * CDIM * CDIM, xbuf, nullptr, nullptr,
            ada_l + 2 * 768, CT, CDIM, CDIM, 4608);
        // h2 = ln(x)*(1+scp)+sp -> hbuf (bf16)
        ln_mod_kernel<<<CT, 256, 0, stream>>>(xbuf, norm2_w + l * CDIM, ada_l,
                                              2304, 3072, 4608, hbuf);
        // m1 = gelu(h2 @ mlp_w1 + b1) -> m1buf (bf16)
        gemm_bf16_kernel<1, true><<<dim3(CMLP / 128, CT / 128), 256, 0, stream>>>(
            hbuf, mlp_w1 + (size_t)l * CDIM * CMLP, nullptr, m1buf,
            mlp_b1 + (size_t)l * CMLP, nullptr, CT, CMLP, CDIM, 0);
        // x += gp * (m1 @ mlp_w2 + b2)
        gemm_bf16_kernel<2, true><<<dim3(CDIM / 128, CT / 128), 256, 0, stream>>>(
            m1buf, mlp_w2 + (size_t)l * CMLP * CDIM, xbuf, nullptr,
            mlp_b2 + (size_t)l * CDIM, ada_l + 5 * 768, CT, CDIM, CMLP, 4608);
    }

    // final LN + modulation -> hbuf (bf16)
    ln_mod_kernel<<<CT, 256, 0, stream>>>(xbuf, fin_norm, finada, 0, 768, 1536, hbuf);
    // logits = hbuf @ fin_w + fin_b
    gemm_bf16_kernel<0, true><<<dim3(CVOC / 128, CT / 128), 256, 0, stream>>>(
        hbuf, fin_w, out, nullptr, fin_b, nullptr, CT, CVOC, CDIM, 0);
}
